// TemporalAttention_71201967833666
// MI455X (gfx1250) — compile-verified, matrix operand order checked by reading, not run on hardware
//
#include <hip/hip_runtime.h>

// ---------------------------------------------------------------------------
// Single-head attention, B=4, S=4096, H=128, fp32 in/out.
// bf16 WMMA (v_wmma_f32_16x16x32_bf16) + fp32 accum, flash-attention style.
// Both attention GEMMs are computed transposed (C = S^T, C = O^T = V^T P^T):
//  - softmax'd P registers serve directly as the next WMMA operand (no LDS)
//  - softmax stats, alpha rescale and 1/l normalize are scalar per lane
//  - every output store is a packed 16B/32B row store.
// ---------------------------------------------------------------------------

typedef __attribute__((ext_vector_type(16))) __bf16 v16bf;
typedef __attribute__((ext_vector_type(8)))  __bf16 v8bf;
typedef __attribute__((ext_vector_type(4)))  __bf16 v4bf;
typedef __attribute__((ext_vector_type(8)))  float  v8f;
typedef __attribute__((ext_vector_type(4)))  float  v4f;

#define SEQ   4096
#define HDIM  128
#define NBAT  4
#define TPAD  144   // padded LDS stride for 128-wide tiles (288 B, 16B-aligned)
#define VPAD  80    // padded LDS stride for 64-wide tiles  (160 B, 16B-aligned)

// 16-bit fragment, per ISA 7.12.2: lane owns row/col (lane&15); its 16 elements
// are two contiguous 8-element runs at K = h8, h8+16 with h8 = 8*(lane>=16).
// => two 16-byte loads (ds_load_b128 / global_load_b128).
__device__ __forceinline__ v16bf load_frag_row(const __bf16* base, int stride, int lane) {
  const __bf16* rowp = base + (lane & 15) * stride + ((lane >> 4) << 3);
  v8bf lo = *(const v8bf*)(rowp);
  v8bf hi = *(const v8bf*)(rowp + 16);
  return __builtin_shufflevector(lo, hi, 0, 1, 2, 3, 4, 5, 6, 7,
                                 8, 9, 10, 11, 12, 13, 14, 15);
}

__device__ __forceinline__ v8f wmma_bf16(v16bf a, v16bf b, v8f c) {
  return __builtin_amdgcn_wmma_f32_16x16x32_bf16(false, a, false, b, (short)0, c,
                                                 false, false);
}

// ---------------------------------------------------------------------------
// Kernel 1: q,k = x @ W + b (row-major, q pre-scaled by 1/sqrt(H));
//           v written TRANSPOSED: vT[b][h][s].  All packed 16B stores.
// grid: (B*S)/128 blocks of 256 threads (8 waves x 16 rows).
// ---------------------------------------------------------------------------
__global__ __launch_bounds__(256) void qkv_kernel(
    const float* __restrict__ x,
    const float* __restrict__ Wq, const float* __restrict__ bq,
    const float* __restrict__ Wk, const float* __restrict__ bk,
    const float* __restrict__ Wv, const float* __restrict__ bv,
    __bf16* __restrict__ q, __bf16* __restrict__ k, __bf16* __restrict__ vT) {
  __shared__ __bf16 xs[128 * TPAD];   // 36 KB, x tile (row-major, padded)
  __shared__ __bf16 wts[HDIM * TPAD]; // 36 KB, W^T (one matrix at a time)

  const int tid  = threadIdx.x;
  const int lane = tid & 31;
  const int wave = tid >> 5;
  const long rowBase = (long)blockIdx.x * 128;

  // stage x tile: float4 loads, 4x bf16 stores
  for (int idx = tid; idx < 128 * 32; idx += 256) {
    const int row = idx >> 5, c = idx & 31;
    v4f xv = *(const v4f*)(x + (rowBase + row) * HDIM + c * 4);
    v4bf o;
#pragma unroll
    for (int j = 0; j < 4; ++j) o[j] = (__bf16)xv[j];
    *(v4bf*)(xs + row * TPAD + c * 4) = o;
  }
  __syncthreads();

  // x fragments for this wave's 16 rows, K=128 (4 chunks of 32)
  v16bf afrag[4];
  {
    const __bf16* xrow = xs + wave * 16 * TPAD;
#pragma unroll
    for (int t = 0; t < 4; ++t) afrag[t] = load_frag_row(xrow + t * 32, TPAD, lane);
  }

  const float qscale = 0.08838834764831845f; // 1/sqrt(128), folded into q
  const int col0 = lane & 15;
  const int rofs = (lane >> 4) << 3; // 0 or 8
  const long bb    = rowBase >> 12;  // batch index
  const long srow0 = (rowBase & (SEQ - 1)) + wave * 16 + rofs;

  // ---- q and k: C = (xW)^T = W^T * x^T -> lane owns one output row ----
  {
    const float* Wmat[2] = {Wq, Wk};
    const float* bias[2] = {bq, bk};
    __bf16*      outm[2] = {q, k};
    for (int m = 0; m < 2; ++m) {
      __syncthreads();
      for (int idx = tid; idx < HDIM * HDIM; idx += 256) {
        const int kk = idx >> 7, nn = idx & 127;
        wts[nn * TPAD + kk] = (__bf16)Wmat[m][idx];
      }
      __syncthreads();

      const float sc = (m == 0) ? qscale : 1.0f;
      __bf16* orow = outm[m] + (rowBase + wave * 16 + col0) * HDIM;
#pragma unroll
      for (int nt = 0; nt < 8; ++nt) {
        v8f acc = {};
#pragma unroll
        for (int t = 0; t < 4; ++t) {
          v16bf wfrag = load_frag_row(wts + (nt * 16) * TPAD + t * 32, TPAD, lane);
          acc = wmma_bf16(wfrag, afrag[t], acc); // C = W^T x^T
        }
        // lane's columns nt*16+rofs .. +7 are contiguous; bias is lane-uniform
        v8bf pk;
#pragma unroll
        for (int r = 0; r < 8; ++r)
          pk[r] = (__bf16)((acc[r] + bias[m][nt * 16 + rofs + r]) * sc);
        *(v8bf*)(orow + nt * 16 + rofs) = pk;
      }
    }
  }

  // ---- v: C = xW (lane owns column; rows contiguous -> packed vT store) ----
  {
    __syncthreads();
    for (int idx = tid; idx < HDIM * HDIM; idx += 256) {
      const int kk = idx >> 7, nn = idx & 127;
      wts[nn * TPAD + kk] = (__bf16)Wv[idx];
    }
    __syncthreads();
#pragma unroll
    for (int nt = 0; nt < 8; ++nt) {
      v8f acc = {};
#pragma unroll
      for (int t = 0; t < 4; ++t) {
        v16bf wfrag = load_frag_row(wts + (nt * 16) * TPAD + t * 32, TPAD, lane);
        acc = wmma_bf16(afrag[t], wfrag, acc); // C = x W
      }
      const int col  = nt * 16 + col0;
      const float bi = bv[col];
      v8bf pk;
#pragma unroll
      for (int r = 0; r < 8; ++r) pk[r] = (__bf16)(acc[r] + bi);
      *(v8bf*)(vT + (bb * HDIM + col) * SEQ + srow0) = pk;
    }
  }
}

// ---------------------------------------------------------------------------
// Kernel 2: flash attention. grid (S/128, B), block 256 (8 waves x 16 q-rows).
// C = S^T then C = O^T; lane owns one query end-to-end.
// ---------------------------------------------------------------------------
__global__ __launch_bounds__(256) void attn_kernel(
    const __bf16* __restrict__ Q, const __bf16* __restrict__ K,
    const __bf16* __restrict__ vT, __bf16* __restrict__ O) {
  __shared__ __bf16 ks[64 * TPAD];    // 18 KB, K block rows (padded)
  __shared__ __bf16 vts[HDIM * VPAD]; // 20 KB, V^T block (padded)

  const int tid  = threadIdx.x;
  const int lane = tid & 31;
  const int wave = tid >> 5;
  const int b    = blockIdx.y;
  const long qBase = (long)b * SEQ + (long)blockIdx.x * 128;

  const __bf16* Qb  = Q + qBase * HDIM;
  const __bf16* Kb  = K + (long)b * SEQ * HDIM;
  const __bf16* vTb = vT + (long)b * HDIM * SEQ;

  // Q fragments: B operand of S^T (lane owns query = lane&15); b128 loads
  v16bf qfrag[4];
  {
    const __bf16* qrow = Qb + wave * 16 * HDIM;
#pragma unroll
    for (int t = 0; t < 4; ++t) qfrag[t] = load_frag_row(qrow + t * 32, HDIM, lane);
  }

  v8f acc[8]; // O^T accumulator: lane owns query (lane&15); reg r -> h
#pragma unroll
  for (int nt = 0; nt < 8; ++nt) acc[nt] = (v8f){};
  float mi = -1e30f, li = 0.0f; // per-lane stats for query (lane&15)

  const int colb = lane & 15;
  const int rofs = (lane >> 4) << 3;

  for (int kb = 0; kb < SEQ; kb += 64) {
    __syncthreads(); // everyone done reading previous ks/vts
    // K rows: 16-byte chunked copy (64 rows x 16 chunks)
    for (int idx = tid; idx < 64 * 16; idx += 256) {
      const int row = idx >> 4, c = idx & 15;
      *(v8bf*)(ks + row * TPAD + c * 8) =
          *(const v8bf*)(Kb + (long)(kb + row) * HDIM + c * 8);
    }
    // V^T block: 16-byte chunked copy from transposed global vT
    for (int idx = tid; idx < HDIM * 8; idx += 256) {
      const int h = idx >> 3, c = idx & 7;
      *(v8bf*)(vts + h * VPAD + c * 8) =
          *(const v8bf*)(vTb + (long)h * SEQ + kb + c * 8);
    }
    __syncthreads();

    // S^T tiles: A = K rows, B = Q (fixed across kt for each t).
    v8f sc[4];
#pragma unroll
    for (int kt = 0; kt < 4; ++kt) sc[kt] = (v8f){};
#pragma unroll
    for (int t = 0; t < 4; ++t)
#pragma unroll
      for (int kt = 0; kt < 4; ++kt) {
        v16bf kf = load_frag_row(ks + (kt * 16) * TPAD + t * 32, TPAD, lane);
        sc[kt] = wmma_bf16(kf, qfrag[t], sc[kt]);
      }

    // per-lane online softmax (32 of 64 keys in-lane; merge halves with 1 shfl)
    float mx = sc[0][0];
#pragma unroll
    for (int kt = 0; kt < 4; ++kt)
#pragma unroll
      for (int r = 0; r < 8; ++r) mx = fmaxf(mx, sc[kt][r]);
    mx = fmaxf(mx, __shfl_xor(mx, 16, 32));
    const float mnew  = fmaxf(mi, mx);
    const float alpha = __expf(mi - mnew);
    mi = mnew;

    float rs = 0.0f;
#pragma unroll
    for (int kt = 0; kt < 4; ++kt)
#pragma unroll
      for (int r = 0; r < 8; ++r) {
        const float p = __expf(sc[kt][r] - mnew);
        sc[kt][r] = p;
        rs += p;
      }
    rs += __shfl_xor(rs, 16, 32);
    li = li * alpha + rs;

    // P^T operand directly from S^T C-registers (pure register packing);
    // serves as the B operand of O^T = V^T P^T.
    v16bf pf[2];
#pragma unroll
    for (int t = 0; t < 2; ++t)
#pragma unroll
      for (int i = 0; i < 8; ++i) {
        pf[t][i]     = (__bf16)sc[2 * t][i];
        pf[t][8 + i] = (__bf16)sc[2 * t + 1][i];
      }

    // scalar rescale (alpha is this lane's query's factor)
#pragma unroll
    for (int nt = 0; nt < 8; ++nt)
#pragma unroll
      for (int r = 0; r < 8; ++r) acc[nt][r] *= alpha;

    // O^T += V^T (A, from vts) x P^T (B, fixed across nt)
#pragma unroll
    for (int t = 0; t < 2; ++t)
#pragma unroll
      for (int nt = 0; nt < 8; ++nt) {
        v16bf vf = load_frag_row(vts + (nt * 16) * VPAD + t * 32, VPAD, lane);
        acc[nt] = wmma_bf16(vf, pf[t], acc[nt]);
      }
  }

  // normalize and emit: lane owns one attended row; packed 16B stores
  {
    const float inv = 1.0f / li;
    __bf16* orow = O + (qBase + wave * 16 + colb) * HDIM;
#pragma unroll
    for (int nt = 0; nt < 8; ++nt) {
      v8bf pk;
#pragma unroll
      for (int r = 0; r < 8; ++r) pk[r] = (__bf16)(acc[nt][r] * inv);
      *(v8bf*)(orow + nt * 16 + rofs) = pk;
    }
  }
}

// ---------------------------------------------------------------------------
// Kernel 3: out = attended @ Wo + bo (fp32), C = (A Wo)^T = Wo^T A^T
// -> lane owns one output row; packed 32B stores; lane-uniform bias.
// ---------------------------------------------------------------------------
__global__ __launch_bounds__(256) void oproj_kernel(
    const __bf16* __restrict__ A, const float* __restrict__ Wo,
    const float* __restrict__ bo, float* __restrict__ out) {
  __shared__ __bf16 as_[128 * TPAD];  // attended tile (row-major, padded)
  __shared__ __bf16 wts[HDIM * TPAD]; // Wo^T

  const int tid  = threadIdx.x;
  const int lane = tid & 31;
  const int wave = tid >> 5;
  const long rowBase = (long)blockIdx.x * 128;

  for (int idx = tid; idx < 128 * 16; idx += 256) {
    const int row = idx >> 4, c = idx & 15;
    *(v8bf*)(as_ + row * TPAD + c * 8) =
        *(const v8bf*)(A + (rowBase + row) * HDIM + c * 8);
  }
  for (int idx = tid; idx < HDIM * HDIM; idx += 256) {
    const int kk = idx >> 7, nn = idx & 127;
    wts[nn * TPAD + kk] = (__bf16)Wo[idx];
  }
  __syncthreads();

  v16bf afrag[4];
  {
    const __bf16* arow = as_ + wave * 16 * TPAD;
#pragma unroll
    for (int t = 0; t < 4; ++t) afrag[t] = load_frag_row(arow + t * 32, TPAD, lane);
  }

  const int col0 = lane & 15;
  const int rofs = (lane >> 4) << 3;
  float* orow = out + (rowBase + wave * 16 + col0) * HDIM;

#pragma unroll
  for (int nt = 0; nt < 8; ++nt) {
    v8f acc = {};
#pragma unroll
    for (int t = 0; t < 4; ++t) {
      v16bf wfrag = load_frag_row(wts + (nt * 16) * TPAD + t * 32, TPAD, lane);
      acc = wmma_bf16(wfrag, afrag[t], acc); // C = Wo^T A^T
    }
    v8f pk;
#pragma unroll
    for (int r = 0; r < 8; ++r)
      pk[r] = acc[r] + bo[nt * 16 + rofs + r]; // lane-uniform bias (SGPR)
    *(v8f*)(orow + nt * 16 + rofs) = pk;
  }
}

// ---------------------------------------------------------------------------
extern "C" void kernel_launch(void* const* d_in, const int* in_sizes, int n_in,
                              void* d_out, int out_size, void* d_ws, size_t ws_size,
                              hipStream_t stream) {
  const float* x  = (const float*)d_in[0];
  const float* Wq = (const float*)d_in[1];
  const float* bq = (const float*)d_in[2];
  const float* Wk = (const float*)d_in[3];
  const float* bk = (const float*)d_in[4];
  const float* Wv = (const float*)d_in[5];
  const float* bv = (const float*)d_in[6];
  const float* Wo = (const float*)d_in[7];
  const float* bo = (const float*)d_in[8];
  float* out = (float*)d_out;

  const size_t n = (size_t)NBAT * SEQ * HDIM; // 2,097,152 elements
  __bf16* q   = (__bf16*)d_ws;
  __bf16* k   = q + n;
  __bf16* vT  = k + n;  // transposed: [B][H][S]
  __bf16* att = vT + n; // total 16 MB bf16 workspace

  const int rowBlocks = (NBAT * SEQ) / 128; // 128

  qkv_kernel<<<rowBlocks, 256, 0, stream>>>(x, Wq, bq, Wk, bk, Wv, bv, q, k, vT);

  dim3 g2(SEQ / 128, NBAT); // (32, 4)
  attn_kernel<<<g2, 256, 0, stream>>>(q, k, vT, att);

  oproj_kernel<<<rowBlocks, 256, 0, stream>>>(att, Wo, bo, out);
}
